// BFNet_66202625900911
// MI455X (gfx1250) — matrix operand order repaired and checked by reading, not compiled, not run on hardware
//
#include <hip/hip_runtime.h>

// ---------------------------------------------------------------------------
// PauliNet-style wavefunction evaluation for MI455X (gfx1250, wave32, WMMA).
// One block per batch element, 256 threads (8 wave32s).
// All GEMMs mapped to V_WMMA_F32_16X16X32_F16 (K=32 basis dim fits exactly;
// K=64/128 are 2/4 chained WMMAs). F16 inputs / F32 accumulate.
// ---------------------------------------------------------------------------

typedef __attribute__((ext_vector_type(16))) _Float16 v16h;
typedef __attribute__((ext_vector_type(8)))  float    v8f;

#define LOG2F 0.69314718055994530942f

__device__ __forceinline__ float sspf(float x) {
    // shifted softplus, ssp(0)==0
    float sp = (x > 15.f) ? x : log1pf(__expf(x));
    return sp - LOG2F;
}

__device__ __forceinline__ v8f wmma16x16x32(v16h a, v16h b, v8f c) {
    // D = A(16x32,f16) * B(32x16,f16) + C(16x16,f32)
    return __builtin_amdgcn_wmma_f32_16x16x32_f16(
        /*neg_a=*/false, a, /*neg_b=*/false, b,
        /*c_mod=*/(short)0, c, /*reuse_a=*/false, /*reuse_b=*/false);
}

// A fragment (ISA 7.12.2, 16-bit A 16x32): lane<16 holds row m=lane with
// K={0..7,16..23}; lane>=16 holds row m=lane-16 with K={8..15,24..31}.
// p points at (row0, k0) of a row-major f16 LDS matrix with leading dim ld.
__device__ __forceinline__ v16h load_a_lds(const _Float16* p, int ld, int lane) {
    int m  = lane & 15;
    int kh = lane >> 4;
    const _Float16* row = p + m * ld;
    v16h a;
#pragma unroll
    for (int v = 0; v < 8; ++v) {
        int kb = (v < 4) ? (kh * 8 + 2 * v) : (16 + kh * 8 + 2 * (v - 4));
        a[2 * v]     = row[kb];
        a[2 * v + 1] = row[kb + 1];
    }
    return a;
}

// B fragment (32x16): lane<16 holds col n=lane, K={0..7,16..23} (2 per VGPR);
// lane>=16 holds col n=lane-16, K={8..15,24..31}. W points at (k0, n0) of a
// row-major f32 GLOBAL weight matrix with leading dim ld; converted to f16.
__device__ __forceinline__ v16h load_b_gf32(const float* W, int ld, int lane) {
    int n  = lane & 15;
    int kh = lane >> 4;
    v16h b;
#pragma unroll
    for (int v = 0; v < 8; ++v) {
        int k = (v < 4) ? (kh * 8 + 2 * v) : (16 + kh * 8 + 2 * (v - 4));
        b[2 * v]     = (_Float16)W[k * ld + n];
        b[2 * v + 1] = (_Float16)W[(k + 1) * ld + n];
    }
    return b;
}

// 8x8 determinant via partially-pivoted Gaussian elimination (single lane).
__device__ float det8(const float* m) {
    float a[64];
#pragma unroll
    for (int i = 0; i < 64; ++i) a[i] = m[i];
    float det = 1.f;
    for (int c = 0; c < 8; ++c) {
        int p = c; float mx = fabsf(a[c * 8 + c]);
        for (int r = c + 1; r < 8; ++r) {
            float v = fabsf(a[r * 8 + c]);
            if (v > mx) { mx = v; p = r; }
        }
        if (p != c) {
            for (int k = 0; k < 8; ++k) {
                float t = a[c * 8 + k]; a[c * 8 + k] = a[p * 8 + k]; a[p * 8 + k] = t;
            }
            det = -det;
        }
        float piv = a[c * 8 + c];
        if (piv == 0.f) return 0.f;
        det *= piv;
        float inv = 1.f / piv;
        for (int r = c + 1; r < 8; ++r) {
            float f = a[r * 8 + c] * inv;
            for (int k = c + 1; k < 8; ++k) a[r * 8 + k] -= f * a[c * 8 + k];
        }
    }
    return det;
}

#define N_EL 16
#define N_UP 8
#define N_NUC 4
#define BASIS 32
#define KD 64
#define ED 128

// LDS layout (total 65280 B < 64 KB)
#define SM_DBH   0        // _Float16[320*32]  Gaussian basis (A-matrix)
#define SM_WSC   20480    // _Float16[8][16*64] per-wave WMMA scratch
#define SM_XS    36864    // float[16*128]     electron embeddings
#define SM_XSH   45056    // _Float16[16*128]
#define SM_ZS    49152    // float[20*64]
#define SM_MSG   54272    // float[16*64]
#define SM_MSGH  58368    // _Float16[16*64]
#define SM_H2H   60416    // _Float16[16*128]
#define SM_DNUC  64512    // float[16*4]
#define SM_SLM   64768    // float[2][8][8]

__global__ __launch_bounds__(256)
void paulinet_kernel(const float* __restrict__ rs,       // [B,16,3]
                     const float* __restrict__ coords,   // [4,3]
                     const float* __restrict__ charges,  // [4]
                     const float* __restrict__ emb_elec, // [16,128]
                     const float* __restrict__ emb_nuc,  // [4,64]
                     const float* __restrict__ kW1,      // [3,32,64]
                     const float* __restrict__ kb1,      // [3,64]
                     const float* __restrict__ kW2,      // [3,64,64]
                     const float* __restrict__ kb2,      // [3,64]
                     const float* __restrict__ einW,     // [3,128,64]
                     const float* __restrict__ eoW1,     // [3,64,128]
                     const float* __restrict__ eob1,     // [3,128]
                     const float* __restrict__ eoW2,     // [3,128,128]
                     const float* __restrict__ eob2,     // [3,128]
                     const float* __restrict__ oW1,      // [16,128,25]
                     const float* __restrict__ ob1,      // [16,25]
                     const float* __restrict__ oW2,      // [16,25,5]
                     const float* __restrict__ ob2,      // [16,5]
                     const float* __restrict__ oW3,      // [16,5,1]
                     const float* __restrict__ ob3,      // [16,1]
                     float* __restrict__ out)            // [B]
{
    __shared__ unsigned char smem[65280] __attribute__((aligned(16)));
    _Float16* dbh  = (_Float16*)(smem + SM_DBH);
    float*    xs   = (float*)   (smem + SM_XS);
    _Float16* xsh  = (_Float16*)(smem + SM_XSH);
    float*    zs   = (float*)   (smem + SM_ZS);
    float*    msg  = (float*)   (smem + SM_MSG);
    _Float16* msgh = (_Float16*)(smem + SM_MSGH);
    _Float16* h2h  = (_Float16*)(smem + SM_H2H);
    float*    dnuc = (float*)   (smem + SM_DNUC);
    float*    slm  = (float*)   (smem + SM_SLM);

    const int b    = blockIdx.x;
    const int tid  = threadIdx.x;
    const int lane = tid & 31;
    const int wave = tid >> 5;

    // -------------------- Phase 0: distances + Gaussian basis --------------
    const float* rsb = rs + (size_t)b * N_EL * 3;
    for (int p = tid; p < N_EL * 20; p += 256) {
        int i = p / 20, j = p - i * 20;
        float ax = rsb[i * 3], ay = rsb[i * 3 + 1], az = rsb[i * 3 + 2];
        float bx, by, bz;
        if (j < N_EL) { bx = rsb[j * 3]; by = rsb[j * 3 + 1]; bz = rsb[j * 3 + 2]; }
        else { int nn = j - N_EL; bx = coords[nn * 3]; by = coords[nn * 3 + 1]; bz = coords[nn * 3 + 2]; }
        float dx = ax - bx, dy = ay - by, dz = az - bz;
        float sq = dx * dx + dy * dy + dz * dz;
        float d  = (sq > 0.f) ? sqrtf(sq) : 0.f;
        if (j >= N_EL) dnuc[i * N_NUC + (j - N_EL)] = d;
#pragma unroll
        for (int kq = 0; kq < BASIS; ++kq) {
            float q  = (1.f / 64.f) + (float)kq * ((1.f - 2.f / 64.f) / 31.f);
            float mu = 10.f * q * q;
            float sg = (1.f + 10.f * q) / 7.f;
            float tt = (d - mu) / sg;
            dbh[p * BASIS + kq] = (_Float16)__expf(-tt * tt);
        }
    }
    // xs init = broadcast electron embeddings
    for (int idx = tid; idx < N_EL * ED; idx += 256) xs[idx] = emb_elec[idx];
    __syncthreads();

    // -------------------- Interaction rounds -------------------------------
    for (int t = 0; t < 3; ++t) {
        const float* kW1t  = kW1  + t * BASIS * KD;
        const float* kb1t  = kb1  + t * KD;
        const float* kW2t  = kW2  + t * KD * KD;
        const float* kb2t  = kb2  + t * KD;
        const float* einWt = einW + t * ED * KD;
        const float* eoW1t = eoW1 + t * KD * ED;
        const float* eob1t = eob1 + t * ED;
        const float* eoW2t = eoW2 + t * ED * ED;
        const float* eob2t = eob2 + t * ED;

        // xs -> f16
        for (int idx = tid; idx < N_EL * ED; idx += 256)
            xsh[idx] = (_Float16)xs[idx];
        __syncthreads();

        // zs = xs @ einW (waves 0-3: one 16-col tile each, K=128 -> 4 WMMA)
        if (wave < 4) {
            int n0 = wave * 16;
            v8f c = {};
#pragma unroll
            for (int ks = 0; ks < 4; ++ks) {
                v16h a  = load_a_lds(xsh + ks * 32, ED, lane);
                v16h bm = load_b_gf32(einWt + (ks * 32) * KD + n0, KD, lane);
                c = wmma16x16x32(a, bm, c);
            }
            int rowl = (lane >> 4) * 8, col = n0 + (lane & 15);
#pragma unroll
            for (int r = 0; r < 8; ++r) zs[(rowl + r) * KD + col] = c[r];
        } else if (wave == 4) {
            for (int idx = lane; idx < N_NUC * KD; idx += 32)
                zs[N_EL * KD + idx] = emb_nuc[idx];
        } else if (wave == 5) {
            for (int idx = lane; idx < N_EL * KD; idx += 32) msg[idx] = 0.f;
        }
        __syncthreads();

        // Fused kernel + message: per 16-row tile of the 320 (i,j) pairs:
        //   H1 = ssp(db@kW1 + kb1)  -> per-wave LDS scratch (4 WMMA)
        //   Ws = H1@kW2 + kb2, zero-diag, msg += Ws*zs (8 WMMA + ds_add_f32)
        {
            v16h b1f[4];
#pragma unroll
            for (int n = 0; n < 4; ++n)
                b1f[n] = load_b_gf32(kW1t + n * 16, KD, lane);
            v16h b2f[8];
#pragma unroll
            for (int ks = 0; ks < 2; ++ks)
#pragma unroll
                for (int n = 0; n < 4; ++n)
                    b2f[ks * 4 + n] = load_b_gf32(kW2t + (ks * 32) * KD + n * 16, KD, lane);

            _Float16* wsc = (_Float16*)(smem + SM_WSC) + wave * (16 * KD);
            for (int tile = wave; tile < 20; tile += 8) {
                v16h adb = load_a_lds(dbh + tile * 16 * BASIS, BASIS, lane);
#pragma unroll
                for (int n = 0; n < 4; ++n) {
                    v8f c = {};
                    c = wmma16x16x32(adb, b1f[n], c);
                    int rowl = (lane >> 4) * 8, col = n * 16 + (lane & 15);
                    float bias = kb1t[col];
#pragma unroll
                    for (int r = 0; r < 8; ++r)
                        wsc[(rowl + r) * KD + col] = (_Float16)sspf(c[r] + bias);
                }
                asm volatile("s_wait_dscnt 0" ::: "memory");
                v16h a0 = load_a_lds(wsc,      KD, lane);
                v16h a1 = load_a_lds(wsc + 32, KD, lane);
#pragma unroll
                for (int n = 0; n < 4; ++n) {
                    v8f c = {};
                    c = wmma16x16x32(a0, b2f[n],     c);
                    c = wmma16x16x32(a1, b2f[4 + n], c);
                    int rowl = (lane >> 4) * 8, col = n * 16 + (lane & 15);
                    float bias = kb2t[col];
#pragma unroll
                    for (int r = 0; r < 8; ++r) {
                        int flat = tile * 16 + rowl + r;
                        int i = flat / 20, j = flat - i * 20;
                        if (i != j) {  // ZeroDiagKernel (diag only exists for j<16)
                            float wv = c[r] + bias;
                            atomicAdd(&msg[i * KD + col], wv * zs[j * KD + col]);
                        }
                    }
                }
            }
        }
        __syncthreads();

        // msg -> f16
        for (int idx = tid; idx < N_EL * KD; idx += 256)
            msgh[idx] = (_Float16)msg[idx];
        __syncthreads();

        // h2 = ssp(msg @ eoW1 + eob1): wave w owns col tile w (2 WMMA)
        {
            v16h a0 = load_a_lds(msgh,      KD, lane);
            v16h a1 = load_a_lds(msgh + 32, KD, lane);
            v8f c = {};
            c = wmma16x16x32(a0, load_b_gf32(eoW1t + wave * 16,            ED, lane), c);
            c = wmma16x16x32(a1, load_b_gf32(eoW1t + 32 * ED + wave * 16,  ED, lane), c);
            int rowl = (lane >> 4) * 8, col = wave * 16 + (lane & 15);
            float bias = eob1t[col];
#pragma unroll
            for (int r = 0; r < 8; ++r)
                h2h[(rowl + r) * ED + col] = (_Float16)sspf(c[r] + bias);
        }
        __syncthreads();

        // xs += h2 @ eoW2 + eob2: wave w owns col tile w (4 WMMA)
        {
            v8f c = {};
#pragma unroll
            for (int ks = 0; ks < 4; ++ks) {
                v16h a  = load_a_lds(h2h + ks * 32, ED, lane);
                v16h bm = load_b_gf32(eoW2t + (ks * 32) * ED + wave * 16, ED, lane);
                c = wmma16x16x32(a, bm, c);
            }
            int rowl = (lane >> 4) * 8, col = wave * 16 + (lane & 15);
            float bias = eob2t[col];
#pragma unroll
            for (int r = 0; r < 8; ++r)
                xs[(rowl + r) * ED + col] += c[r] + bias;
        }
        __syncthreads();
    }

    // -------------------- Slater orbital nets (VALU; tiny, ragged) ---------
    float* wf = (float*)(smem + SM_WSC) + wave * 512;  // per-wave f32 scratch
    for (int rr = 0; rr < 2; ++rr) {
        int o  = wave + rr * 8;        // orbital 0..15
        int eb = (o < N_UP) ? 0 : N_UP;
        // layer 1: lanes 0..24 own hidden unit h; 8 electrons each
        if (lane < 25) {
            float acc[8];
#pragma unroll
            for (int e = 0; e < 8; ++e) acc[e] = 0.f;
            const float* W1 = oW1 + o * (ED * 25) + lane;
            for (int k = 0; k < ED; ++k) {
                float wv = W1[k * 25];
#pragma unroll
                for (int e = 0; e < 8; ++e) acc[e] += xs[(eb + e) * ED + k] * wv;
            }
            float bv = ob1[o * 25 + lane];
#pragma unroll
            for (int e = 0; e < 8; ++e) wf[e * 25 + lane] = sspf(acc[e] + bv);
        }
        asm volatile("s_wait_dscnt 0" ::: "memory");
        // layer 2: 40 outputs (e,g) over lanes
        for (int pass = 0; pass < 2; ++pass) {
            int idx = lane + pass * 32;
            if (idx < 40) {
                int e = idx / 5, g = idx - e * 5;
                float acc = ob2[o * 5 + g];
                for (int k = 0; k < 25; ++k)
                    acc += wf[e * 25 + k] * oW2[o * 125 + k * 5 + g];
                wf[200 + idx] = sspf(acc);
            }
        }
        asm volatile("s_wait_dscnt 0" ::: "memory");
        // layer 3: lanes 0..7 own electron e -> slater matrix entry
        if (lane < 8) {
            float acc = ob3[o];
#pragma unroll
            for (int g = 0; g < 5; ++g)
                acc += wf[200 + lane * 5 + g] * oW3[o * 5 + g];
            int spin = (o < N_UP) ? 0 : 1;
            slm[spin * 64 + lane * 8 + (o & 7)] = acc;  // [elec, orb]
        }
        asm volatile("s_wait_dscnt 0" ::: "memory");
    }
    __syncthreads();

    // -------------------- Determinants + nuclear asymptotic ----------------
    if (tid == 0) {
        float det_u = det8(slm);
        float det_d = det8(slm + 64);
        float prod = 1.f;
        for (int e = 0; e < N_EL; ++e) {
            float s = 0.f;
            for (int n = 0; n < N_NUC; ++n) {
                float d = dnuc[e * N_NUC + n];
                // decay = sqrt(2*0.5) = 1, alpha = 1
                s += __expf(-(charges[n] * d + d * d) / (1.f + d));
            }
            prod *= s;
        }
        out[b] = det_u * det_d * prod;
    }
}

extern "C" void kernel_launch(void* const* d_in, const int* in_sizes, int n_in,
                              void* d_out, int out_size, void* d_ws, size_t ws_size,
                              hipStream_t stream) {
    (void)n_in; (void)d_ws; (void)ws_size; (void)out_size;
    const float* rs       = (const float*)d_in[0];
    const float* coords   = (const float*)d_in[1];
    const float* charges  = (const float*)d_in[2];
    const float* emb_elec = (const float*)d_in[3];
    const float* emb_nuc  = (const float*)d_in[4];
    const float* kW1      = (const float*)d_in[5];
    const float* kb1      = (const float*)d_in[6];
    const float* kW2      = (const float*)d_in[7];
    const float* kb2      = (const float*)d_in[8];
    const float* einW     = (const float*)d_in[9];
    const float* eoW1     = (const float*)d_in[10];
    const float* eob1     = (const float*)d_in[11];
    const float* eoW2     = (const float*)d_in[12];
    const float* eob2     = (const float*)d_in[13];
    const float* oW1      = (const float*)d_in[14];
    const float* ob1      = (const float*)d_in[15];
    const float* oW2      = (const float*)d_in[16];
    const float* ob2      = (const float*)d_in[17];
    const float* oW3      = (const float*)d_in[18];
    const float* ob3      = (const float*)d_in[19];

    int B = in_sizes[0] / (N_EL * 3);
    paulinet_kernel<<<dim3(B), dim3(256), 0, stream>>>(
        rs, coords, charges, emb_elec, emb_nuc,
        kW1, kb1, kW2, kb2, einW, eoW1, eob1, eoW2, eob2,
        oW1, ob1, oW2, ob2, oW3, ob3, (float*)d_out);
}
